// TransformerBlock_52896817217711
// MI455X (gfx1250) — matrix operand order repaired and checked by reading, not compiled, not run on hardware
//
#include <hip/hip_runtime.h>
#include <math.h>

// ---------------- problem constants ----------------
#define B_   2
#define S_   512
#define P_   79      // prev nodes
#define H_   12      // heads
#define D_   768     // d_model
#define DH_  64      // d_head
#define DM_  3072    // d_mlp
#define PH_  91      // P_ + H_
#define ON_  92      // output nodes per token
#define NT_  (B_*S_) // 1024 tokens
#define KC_  32      // GEMM k-chunk

typedef __bf16 bf16x8  __attribute__((ext_vector_type(8)));
typedef __bf16 bf16x16 __attribute__((ext_vector_type(16)));
typedef float  f32x8   __attribute__((ext_vector_type(8)));

// 64-bit payload type for the async-LDS builtin (as revealed by the compiler:
// param is "vector_size(8) int" pointer in AS1 / AS3).
typedef int v2i_vs __attribute__((vector_size(8)));
typedef __attribute__((address_space(1))) v2i_vs as1_v2i;
typedef __attribute__((address_space(3))) v2i_vs as3_v2i;

// Detect gfx1250 async global->LDS builtins (device pass only; host falls back).
#if defined(__has_builtin)
#if __has_builtin(__builtin_amdgcn_global_load_async_to_lds_b64) && \
    __has_builtin(__builtin_amdgcn_s_wait_asynccnt)
#define HAVE_ASYNC_LDS 1
#endif
#endif
#ifndef HAVE_ASYNC_LDS
#define HAVE_ASYNC_LDS 0
#endif

// ---------------- helpers ----------------
__device__ __forceinline__ float block_reduce_sum(float v, float* buf) {
    int t = threadIdx.x;
    buf[t] = v;
    __syncthreads();
    for (int s = blockDim.x >> 1; s > 0; s >>= 1) {
        if (t < s) buf[t] += buf[t + s];
        __syncthreads();
    }
    float r = buf[0];
    __syncthreads();
    return r;
}

__device__ __forceinline__ bf16x16 load_frag16(const __bf16* p0, const __bf16* p1) {
    bf16x8 lo = *(const bf16x8*)p0;
    bf16x8 hi = *(const bf16x8*)p1;
    return __builtin_shufflevector(lo, hi, 0,1,2,3,4,5,6,7,8,9,10,11,12,13,14,15);
}

// ---------------- kernels ----------------
// Transposing fp32 -> bf16 weight conversion: src is [slabs][K][N] row-major,
// dst is [slabs][N][K] row-major (B^T layout for WMMA B fragments).
__global__ void cvt_t_kernel(const float* __restrict__ s, __bf16* __restrict__ d,
                             long K, long N, long total) {
    long i = (long)blockIdx.x * blockDim.x + threadIdx.x;
    if (i >= total) return;
    long kn   = K * N;
    long slab = i / kn;
    long r    = i - slab * kn;
    long n    = r / K;
    long kk   = r - n * K;
    d[i] = (__bf16)s[slab * kn + kk * N + n];
}

__global__ void copy_resid_kernel(const float4* __restrict__ src, float* __restrict__ out) {
    long i = (long)blockIdx.x * blockDim.x + threadIdx.x;   // float4 index
    const long TOT = (long)B_ * S_ * P_ * D_ / 4;
    if (i >= TOT) return;
    const long PV = (long)P_ * (D_ / 4);
    long t  = i / PV;
    long r  = i - t * PV;
    long p  = r / (D_ / 4);
    long dv = r - p * (D_ / 4);
    float4 v = src[i];
    *(float4*)(out + (t * ON_ + p) * D_ + dv * 4) = v;
}

// Per token: masked sum over P prev nodes (mask = logits>0, uniform per block),
// then layernorm over d per head, emit bf16 [H][NT][D].
__global__ __launch_bounds__(256)
void mask_ln_kernel(const float* __restrict__ resid, const float* __restrict__ mlogits,
                    const float* __restrict__ lnw, const float* __restrict__ lnb,
                    __bf16* __restrict__ nout) {
    long t  = blockIdx.x;
    int tid = threadIdx.x;
    __shared__ float    chunk[P_][64];
    __shared__ float    res[H_][D_];
    __shared__ float    rbuf[256];
    __shared__ unsigned mbits[P_];
    if (tid < P_) {
        unsigned m = 0;
        for (int h = 0; h < H_; ++h)
            if (mlogits[tid * H_ + h] > 0.0f) m |= (1u << h);
        mbits[tid] = m;
    }
    __syncthreads();
    const float* rp = resid + t * (long)(P_ * D_);
    for (int c = 0; c < D_ / 64; ++c) {
        for (int idx = tid; idx < P_ * 16; idx += 256) {   // float4-vectorized fill
            int p = idx >> 4, dv = idx & 15;
            *(float4*)&chunk[p][dv * 4] =
                *(const float4*)(rp + (long)p * D_ + c * 64 + dv * 4);
        }
        __syncthreads();
        #pragma unroll
        for (int j = 0; j < 3; ++j) {
            int idx = tid + j * 256;            // 0..767 = h*64 + d
            int h = idx >> 6, d = idx & 63;
            float acc = 0.0f;
            for (int p = 0; p < P_; ++p)
                if ((mbits[p] >> h) & 1u) acc += chunk[p][d];
            res[h][c * 64 + d] = acc;
        }
        __syncthreads();
    }
    for (int h = 0; h < H_; ++h) {
        float s1 = 0.0f, s2 = 0.0f;
        #pragma unroll
        for (int j = 0; j < 3; ++j) {
            float x = res[h][tid + j * 256];
            s1 += x; s2 += x * x;
        }
        s1 = block_reduce_sum(s1, rbuf);
        s2 = block_reduce_sum(s2, rbuf);
        float mu   = s1 * (1.0f / D_);
        float var  = s2 * (1.0f / D_) - mu * mu;
        float rstd = rsqrtf(var + 1e-5f);
        #pragma unroll
        for (int j = 0; j < 3; ++j) {
            int d = tid + j * 256;
            float n = (res[h][d] - mu) * rstd * lnw[d] + lnb[d];
            nout[((long)h * NT_ + t) * D_ + d] = (__bf16)n;
        }
    }
}

// 4-wave WMMA GEMM: block computes a 128x16 C tile. The shared 16xKC B chunk is
// staged into LDS (async global->LDS when available, double-buffered); each wave
// computes 32 rows with software-pipelined direct A fragment loads.
// B operand stored TRANSPOSED ([N][K], leading dim ldbt).
// mode 0: f32 store; 1: bf16 store; 2: exact-erf GELU -> bf16; 3: bf16 transposed store
__global__ __launch_bounds__(128)
void gemm_kernel(const __bf16* __restrict__ A, long lda, long aStrideZ,
                 const __bf16* __restrict__ BT, long ldbt, long bStrideZ,
                 const float* __restrict__ bias, long biasStrideZ, float biasScale,
                 float* __restrict__ Cf, __bf16* __restrict__ Cb, long ldc, long cStrideZ,
                 int K, int mode) {
    __shared__ __bf16 bstage[2][16][KC_];
    long zi = blockIdx.z;
    A  += zi * aStrideZ;
    BT += zi * bStrideZ;
    if (bias) bias += zi * biasStrideZ;
    long row0 = (long)blockIdx.x * 128;
    long col0 = (long)blockIdx.y * 16;
    int tid  = threadIdx.x;
    int wave = tid >> 5;
    int lane = tid & 31;
    int rowA = lane & 15;      // also the B column within the tile
    int hf   = lane >> 4;
    const __bf16* Aw = A + (row0 + wave * 32 + rowA) * lda;

    // B staging: each thread moves 4 bf16 (8B); 128 threads cover 16x32 chunk
    int scol  = tid >> 3;        // 0..15
    int skoff = (tid & 7) * 4;   // 0..28
    const __bf16* bsrc = BT + (col0 + scol) * ldbt + skoff;
    auto stage = [&](int buf, int k0) {
        const __bf16* g = bsrc + k0;
        __bf16* l = &bstage[buf][scol][skoff];
#if HAVE_ASYNC_LDS
        __builtin_amdgcn_global_load_async_to_lds_b64(
            (as1_v2i*)(unsigned long long)(uintptr_t)g,
            (as3_v2i*)(unsigned int)(uintptr_t)l, 0, 0);
#else
        *(unsigned long long*)l = *(const unsigned long long*)g;
#endif
    };

    int nk = K / KC_;
    stage(0, 0);
    // preload A fragments for chunk 0
    bf16x16 a0 = load_frag16(Aw + hf * 8, Aw + hf * 8 + 16);
    bf16x16 a1 = load_frag16(Aw + 16 * lda + hf * 8, Aw + 16 * lda + hf * 8 + 16);
    f32x8 c0 = {}, c1 = {};
    for (int i = 0; i < nk; ++i) {
        int k0 = i * KC_;
        if (i + 1 < nk) stage((i + 1) & 1, k0 + KC_);
#if HAVE_ASYNC_LDS
        if (i + 1 < nk) __builtin_amdgcn_s_wait_asynccnt(1);
        else            __builtin_amdgcn_s_wait_asynccnt(0);
#endif
        __syncthreads();   // B chunk i visible to all waves
        // issue next chunk's A loads before consuming this chunk (pipelining)
        bf16x16 na0 = a0, na1 = a1;
        if (i + 1 < nk) {
            const __bf16* a0p = Aw + k0 + KC_ + hf * 8;
            const __bf16* a1p = a0p + 16 * lda;
            __builtin_prefetch(a0p + KC_, 0, 3);
            na0 = load_frag16(a0p, a0p + 16);
            na1 = load_frag16(a1p, a1p + 16);
        }
        const __bf16* bl = &bstage[i & 1][rowA][hf * 8];
        bf16x16 b = load_frag16(bl, bl + 16);
        c0 = __builtin_amdgcn_wmma_f32_16x16x32_bf16(false, a0, false, b, (short)0, c0,
                                                     false, false);
        c1 = __builtin_amdgcn_wmma_f32_16x16x32_bf16(false, a1, false, b, (short)0, c1,
                                                     false, false);
        a0 = na0;
        a1 = na1;
        __syncthreads();   // all waves done with buffer before it is overwritten
    }

    int col = rowA;
    float bvv = bias ? bias[col0 + col] * biasScale : 0.0f;
    #pragma unroll
    for (int t2 = 0; t2 < 2; ++t2) {
        f32x8 c = t2 ? c1 : c0;
        #pragma unroll
        for (int r = 0; r < 8; ++r) {
            long row = row0 + wave * 32 + t2 * 16 + r + hf * 8;
            float v = c[r] + bvv;
            if (mode == 2) v = 0.5f * v * (1.0f + erff(v * 0.70710678118654752f));
            long zoff = zi * cStrideZ;
            if (mode == 0)      Cf[row * ldc + col0 + col + zoff] = v;
            else if (mode == 3) Cb[(col0 + col) * ldc + row + zoff] = (__bf16)v;
            else                Cb[row * ldc + col0 + col + zoff] = (__bf16)v;
        }
    }
}

// Flash attention, one wave per (b, h, 16-query tile), 32 keys per iteration.
// q/k layout [H][NT][DH] bf16; v is pre-transposed vT [H][DH][NT] bf16.
__global__ __launch_bounds__(32)
void attn_kernel(const __bf16* __restrict__ q, const __bf16* __restrict__ k,
                 const __bf16* __restrict__ vT, __bf16* __restrict__ zb) {
    int bh = blockIdx.y;
    int h  = bh % H_;
    int b  = bh / H_;
    const long slab = ((long)h * NT_ + (long)b * S_) * DH_;
    const __bf16* qp  = q + slab;
    const __bf16* kp  = k + slab;
    const __bf16* vtp = vT + (long)h * DH_ * NT_ + (long)b * S_;
    int qt   = blockIdx.x;
    int qr0  = qt * 16;
    int lane = threadIdx.x & 31;
    int lrow = lane & 15;
    int hf   = lane >> 4;

    // q fragments (dh = 0..31 and 32..63)
    bf16x16 aq[2];
    #pragma unroll
    for (int j = 0; j < 2; ++j) {
        const __bf16* ap = qp + (long)(qr0 + lrow) * DH_ + j * 32 + hf * 8;
        aq[j] = load_frag16(ap, ap + 16);
    }

    f32x8 zc[4];
    #pragma unroll
    for (int d = 0; d < 4; ++d) zc[d] = (f32x8){};
    float m_run[8], l_run[8];
    #pragma unroll
    for (int r = 0; r < 8; ++r) { m_run[r] = -3.0e38f; l_run[r] = 0.0f; }

    __shared__ float pt[16][33];

    int nchunks = qt / 2 + 1;
    for (int c = 0; c < nchunks; ++c) {
        int kc0 = c * 32;
        // scores for 16 queries x 32 keys (two N-tiles, K=64 in two wmma each)
        f32x8 sc[2];
        #pragma unroll
        for (int jn = 0; jn < 2; ++jn) {
            f32x8 s = {};
            #pragma unroll
            for (int j = 0; j < 2; ++j) {
                const __bf16* kpp =
                    kp + (long)(kc0 + jn * 16 + lrow) * DH_ + j * 32 + hf * 8;
                bf16x16 bk = load_frag16(kpp, kpp + 16);
                s = __builtin_amdgcn_wmma_f32_16x16x32_bf16(false, aq[j], false, bk,
                                                            (short)0, s, false, false);
            }
            #pragma unroll
            for (int r = 0; r < 8; ++r) {
                float sv = s[r] * 0.125f;            // 1/sqrt(64)
                int row  = qr0 + r + hf * 8;
                int colg = kc0 + jn * 16 + lrow;
                if (colg > row) sv = -100000.0f;     // IGNORE
                s[r] = sv;
            }
            sc[jn] = s;
        }
        // online softmax over the 32-key chunk (row reductions in 16-lane groups)
        #pragma unroll
        for (int r = 0; r < 8; ++r) {
            float mv = fmaxf(sc[0][r], sc[1][r]);
            #pragma unroll
            for (int msk = 1; msk <= 8; msk <<= 1)
                mv = fmaxf(mv, __shfl_xor(mv, msk, 32));
            float nm   = fmaxf(m_run[r], mv);
            float corr = __expf(m_run[r] - nm);
            m_run[r]   = nm;
            float p0 = __expf(sc[0][r] - nm);
            float p1 = __expf(sc[1][r] - nm);
            float ps = p0 + p1;
            #pragma unroll
            for (int msk = 1; msk <= 8; msk <<= 1)
                ps += __shfl_xor(ps, msk, 32);
            l_run[r] = l_run[r] * corr + ps;
            #pragma unroll
            for (int d = 0; d < 4; ++d) zc[d][r] *= corr;
            sc[0][r] = p0;
            sc[1][r] = p1;
        }
        // re-fragment 16x32 pattern via LDS (C layout -> A layout, full K=32)
        __syncthreads();
        #pragma unroll
        for (int r = 0; r < 8; ++r) {
            pt[r + hf * 8][lrow]      = sc[0][r];
            pt[r + hf * 8][16 + lrow] = sc[1][r];
        }
        __syncthreads();
        bf16x16 apf;
        #pragma unroll
        for (int i = 0; i < 16; ++i) {
            int kk = hf * 16 + i;
            apf[i] = (__bf16)pt[lrow][kk];
        }
        // z += pattern(16x32) @ v(32x64); v fragment from transposed vT, contiguous keys
        #pragma unroll
        for (int d = 0; d < 4; ++d) {
            const __bf16* vp2 = vtp + (long)(d * 16 + lrow) * NT_ + kc0 + hf * 16;
            bf16x16 bv = load_frag16(vp2, vp2 + 8);
            zc[d] = __builtin_amdgcn_wmma_f32_16x16x32_bf16(false, apf, false, bv,
                                                            (short)0, zc[d], false, false);
        }
        __syncthreads();
    }
    // normalize and store z (bf16, [H][NT][DH] layout)
    #pragma unroll
    for (int d = 0; d < 4; ++d)
        #pragma unroll
        for (int r = 0; r < 8; ++r) {
            float val = zc[d][r] / l_run[r];
            zb[slab + (long)(qr0 + r + hf * 8) * DH_ + d * 16 + lrow] = (__bf16)val;
        }
}

// Per token: masked sum over 91 residual nodes (read back from out), LN2 -> bf16.
__global__ __launch_bounds__(256)
void mlp_mask_ln_kernel(const float* __restrict__ outp, const float* __restrict__ mlogits,
                        const float* __restrict__ lnw, const float* __restrict__ lnb,
                        __bf16* __restrict__ nmid) {
    long t  = blockIdx.x;
    int tid = threadIdx.x;
    __shared__ float mm[PH_];
    __shared__ float rbuf[256];
    if (tid < PH_) mm[tid] = (mlogits[tid] > 0.0f) ? 1.0f : 0.0f;
    __syncthreads();
    float x[3];
    float s1 = 0.0f, s2 = 0.0f;
    #pragma unroll
    for (int j = 0; j < 3; ++j) {
        int d = tid + j * 256;
        const float* base = outp + t * (long)(ON_ * D_) + d;
        float acc = 0.0f;
        for (int p = 0; p < PH_; ++p) acc += base[(long)p * D_] * mm[p];
        x[j] = acc;
        s1 += acc; s2 += acc * acc;
    }
    s1 = block_reduce_sum(s1, rbuf);
    s2 = block_reduce_sum(s2, rbuf);
    float mu   = s1 * (1.0f / D_);
    float var  = s2 * (1.0f / D_) - mu * mu;
    float rstd = rsqrtf(var + 1e-5f);
    #pragma unroll
    for (int j = 0; j < 3; ++j) {
        int d = tid + j * 256;
        float n = (x[j] - mu) * rstd * lnw[d] + lnb[d];
        nmid[t * D_ + d] = (__bf16)n;
    }
}

// ---------------- host launcher ----------------
extern "C" void kernel_launch(void* const* d_in, const int* in_sizes, int n_in,
                              void* d_out, int out_size, void* d_ws, size_t ws_size,
                              hipStream_t stream) {
    (void)in_sizes; (void)n_in; (void)out_size; (void)ws_size;
    const float* resid = (const float*)d_in[0];
    const float* mql   = (const float*)d_in[1];
    const float* mkl   = (const float*)d_in[2];
    const float* mvl   = (const float*)d_in[3];
    const float* mmll  = (const float*)d_in[4];
    const float* WQ    = (const float*)d_in[5];
    const float* bQ    = (const float*)d_in[6];
    const float* WK    = (const float*)d_in[7];
    const float* bK    = (const float*)d_in[8];
    const float* WV    = (const float*)d_in[9];
    const float* bV    = (const float*)d_in[10];
    const float* WO    = (const float*)d_in[11];
    const float* bO    = (const float*)d_in[12];
    const float* ln1w  = (const float*)d_in[13];
    const float* ln1b  = (const float*)d_in[14];
    const float* ln2w  = (const float*)d_in[15];
    const float* ln2b  = (const float*)d_in[16];
    const float* Win   = (const float*)d_in[17];
    const float* bin   = (const float*)d_in[18];
    const float* Wout  = (const float*)d_in[19];
    const float* bout  = (const float*)d_in[20];
    float* out = (float*)d_out;
    char*  ws  = (char*)d_ws;

    size_t o = 0;
    auto take = [&](size_t bytes) {
        size_t r = o;
        o += (bytes + 255) & ~(size_t)255;
        return r;
    };
    const size_t WPROJ = (size_t)H_ * D_ * DH_;    // per-head-set weight elems
    __bf16* wqkvT = (__bf16*)(ws + take(3 * WPROJ * 2));             // [3][H][DH][D]
    __bf16* wohT  = (__bf16*)(ws + take((size_t)H_ * D_ * DH_ * 2)); // [H][D][DH]
    __bf16* winT  = (__bf16*)(ws + take((size_t)DM_ * D_ * 2));      // [DM][D]
    __bf16* woutT = (__bf16*)(ws + take((size_t)D_ * DM_ * 2));      // [D][DM]
    __bf16* nbuf  = (__bf16*)(ws + take((size_t)H_ * NT_ * D_ * 2)); // [H][NT][D]
    __bf16* qkv   = (__bf16*)(ws + take(3ul * H_ * NT_ * DH_ * 2));  // q,k:[H][NT][DH]; vT:[H][DH][NT]
    __bf16* zbuf  = (__bf16*)(ws + take((size_t)H_ * NT_ * DH_ * 2));
    __bf16* nmid  = (__bf16*)(ws + take((size_t)NT_ * D_ * 2));
    __bf16* hid   = (__bf16*)(ws + take((size_t)NT_ * DM_ * 2));

    auto cvtT = [&](const float* s, __bf16* d, long K, long N, long total) {
        cvt_t_kernel<<<dim3((unsigned)((total + 255) / 256)), 256, 0, stream>>>(
            s, d, K, N, total);
    };
    // weights stored transposed per slab: dst [slab][N][K]
    cvtT(WQ,   wqkvT,             D_, DH_, (long)H_ * D_ * DH_);
    cvtT(WK,   wqkvT + WPROJ,     D_, DH_, (long)H_ * D_ * DH_);
    cvtT(WV,   wqkvT + 2 * WPROJ, D_, DH_, (long)H_ * D_ * DH_);
    cvtT(WO,   wohT,  DH_, D_, (long)H_ * DH_ * D_);
    cvtT(Win,  winT,  D_, DM_, (long)D_ * DM_);
    cvtT(Wout, woutT, DM_, D_, (long)DM_ * D_);

    // nodes 0..78 of output = resid_pre
    {
        long tot = (long)B_ * S_ * P_ * D_ / 4;
        copy_resid_kernel<<<dim3((unsigned)((tot + 255) / 256)), 256, 0, stream>>>(
            (const float4*)resid, out);
    }

    // q/k/v: masked reduce + LN, then 12-head projection GEMM (WMMA).
    // q,k stored [H][NT][DH] (mode 1); v stored transposed [H][DH][NT] (mode 3).
    const float* mls[3] = { mql, mkl, mvl };
    const float* bqs[3] = { bQ, bK, bV };
    for (int w = 0; w < 3; ++w) {
        mask_ln_kernel<<<dim3(NT_), 256, 0, stream>>>(resid, mls[w], ln1w, ln1b, nbuf);
        int  mode = (w == 2) ? 3 : 1;
        long ldc  = (w == 2) ? (long)NT_ : (long)DH_;
        gemm_kernel<<<dim3(NT_ / 128, DH_ / 16, H_), 128, 0, stream>>>(
            nbuf, D_, (long)NT_ * D_,
            wqkvT + (size_t)w * WPROJ, D_, (long)DH_ * D_,
            bqs[w], DH_, 1.0f,
            nullptr, qkv + (size_t)w * H_ * NT_ * DH_, ldc, (long)NT_ * DH_,
            D_, mode);
    }

    // attention (flash, WMMA, 32 keys/iter)
    attn_kernel<<<dim3(S_ / 16, B_ * H_), 32, 0, stream>>>(
        qkv, qkv + (size_t)H_ * NT_ * DH_, qkv + 2ul * H_ * NT_ * DH_, zbuf);

    // per-head attn_out = z @ W_O[h] + b_O/H, written into out nodes 79..90
    gemm_kernel<<<dim3(NT_ / 128, D_ / 16, H_), 128, 0, stream>>>(
        zbuf, DH_, (long)NT_ * DH_,
        wohT, DH_, (long)D_ * DH_,
        bO, 0, 1.0f / H_,
        out + (size_t)P_ * D_, nullptr, (long)ON_ * D_, D_,
        DH_, 0);

    // MLP mask reduce over 91 nodes + LN2
    mlp_mask_ln_kernel<<<dim3(NT_), 256, 0, stream>>>(out, mmll, ln2w, ln2b, nmid);

    // MLP: 768 -> 3072 with exact GELU (bf16 out), then 3072 -> 768 into node 91
    gemm_kernel<<<dim3(NT_ / 128, DM_ / 16, 1), 128, 0, stream>>>(
        nmid, D_, 0, winT, D_, 0, bin, 0, 1.0f,
        nullptr, hid, DM_, 0, D_, 2);
    gemm_kernel<<<dim3(NT_ / 128, D_ / 16, 1), 128, 0, stream>>>(
        hid, DM_, 0, woutT, DM_, 0, bout, 0, 1.0f,
        out + (size_t)PH_ * D_, nullptr, (long)ON_ * D_, 0,
        DM_, 0);
}
